// Seq2Seq_48215302865040
// MI455X (gfx1250) — matrix operand order repaired and checked by reading, not compiled, not run on hardware
//
#include <hip/hip_runtime.h>

// ---------- types ----------
typedef _Float16 h16;
typedef __attribute__((ext_vector_type(16))) _Float16 v16h;
typedef __attribute__((ext_vector_type(8)))  float    v8f;

union ABu { uint4 u[2]; v16h h; };

#define AA    384   // alphabet / hidden
#define TT    128   // tag dim
#define BB    1024  // batch
#define LL    50
#define START 1

#define WMMA(a, b, c) __builtin_amdgcn_wmma_f32_16x16x32_f16(false, (a), false, (b), (short)0, (c), false, false)

// Fast activations: precision already bounded by the f16 cast of h, so use the
// native v_exp_f32 / v_rcp_f32 path instead of branchy libm expansions.
__device__ __forceinline__ float sigf(float x) {
    return __builtin_amdgcn_rcpf(1.0f + __expf(-x));
}
__device__ __forceinline__ float tanhfast(float x) {
    return 1.0f - 2.0f * __builtin_amdgcn_rcpf(1.0f + __expf(2.0f * x));
}

// ---------------------------------------------------------------------------
// Pack [W_ih | W_hh] -> f16 concatenated weights (row n has K = in_dim + 384
// contiguous halves) and combined bias b_ih + b_hh.
// ---------------------------------------------------------------------------
__global__ __launch_bounds__(256)
void pack_weights(const float* __restrict__ Wih, const float* __restrict__ Whh,
                  const float* __restrict__ bih, const float* __restrict__ bhh,
                  h16* __restrict__ Wcat, float* __restrict__ bsum, int in_dim) {
    const int K = in_dim + AA;
    const int total = 4 * AA * K;
    int tid = blockIdx.x * blockDim.x + threadIdx.x;
    int stride = gridDim.x * blockDim.x;
    for (int i = tid; i < total; i += stride) {
        int n = i / K, k = i - n * K;
        float v = (k < in_dim) ? Wih[(size_t)n * in_dim + k]
                               : Whh[(size_t)n * AA + (k - in_dim)];
        Wcat[i] = (h16)v;
    }
    if (tid < 4 * AA) bsum[tid] = bih[tid] + bhh[tid];
}

__global__ __launch_bounds__(256)
void convert_f16(const float* __restrict__ src, h16* __restrict__ dst, int n) {
    int tid = blockIdx.x * blockDim.x + threadIdx.x;
    int stride = gridDim.x * blockDim.x;
    for (int i = tid; i < n; i += stride) dst[i] = (h16)src[i];
}

// ---------------------------------------------------------------------------
// Initialize activation buffers, cell state, and out[:,0,:] (start one-hot).
// X0 row (1280 halves): [prev 0:384 | tags 384:512 | h0_ping 512:896 | h0_pong 896:1280]
// X1/X2 row (1152 halves): [x_in 0:384 | h_ping 384:768 | h_pong 768:1152]
// ---------------------------------------------------------------------------
__global__ __launch_bounds__(256)
void init_state(const float* __restrict__ tags,
                h16* __restrict__ X0, h16* __restrict__ X1, h16* __restrict__ X2,
                h16* __restrict__ Hlin, float* __restrict__ cst, float* __restrict__ out) {
    int tid = blockIdx.x * blockDim.x + threadIdx.x;
    int stride = gridDim.x * blockDim.x;
    for (int i = tid; i < BB * 1280; i += stride) {
        int b = i / 1280, col = i - b * 1280;
        h16 v = (h16)0.0f;
        if (col == START) v = (h16)1.0f;
        else if (col >= AA && col < AA + TT) v = (h16)tags[(size_t)b * TT + (col - AA)];
        X0[i] = v;
    }
    for (int i = tid; i < BB * 1152; i += stride) { X1[i] = (h16)0.0f; X2[i] = (h16)0.0f; }
    for (int i = tid; i < BB * AA;   i += stride) Hlin[i] = (h16)0.0f;
    for (int i = tid; i < 3 * BB * AA; i += stride) cst[i] = 0.0f;
    for (int i = tid; i < BB * AA; i += stride) {
        int b = i / AA, a = i - b * AA;
        out[(size_t)b * LL * AA + a] = (a == START) ? 1.0f : 0.0f;   // t = 0 slice
    }
}

// ---------------------------------------------------------------------------
// One LSTM layer, one timestep. One wave owns a 32(batch) x 16(unit) tile and
// ALL FOUR gates for it: 8 f32 accumulators, 8 WMMAs per K-step reusing 2
// A-frags and 4 B-frags. Pointwise update runs entirely in registers (the C/D
// layout keeps all gates of element (m,n) in the same lane & VGPR index).
// Grid: 32 mtiles * 24 ntiles = 768 waves = 192 blocks of 128 threads.
// ---------------------------------------------------------------------------
__global__ __launch_bounds__(128)
void lstm_layer(h16* __restrict__ X, int ldx, int in_dim,
                int h_read_off, int h_write_off,
                const h16* __restrict__ Wcat, const float* __restrict__ bsum,
                float* __restrict__ cst,
                h16* __restrict__ Xnext, int ld_next) {
    const int lane = threadIdx.x & 31;
    const int wid  = blockIdx.x * 4 + (threadIdx.x >> 5);   // 0..767
    const int m0 = (wid / 24) * 32;
    const int n0 = (wid % 24) * 16;
    const int K  = in_dim + AA;

    const int mrow  = lane & 15;
    const int ksel  = (lane & 16) ? 8  : 0;     // A-frag K select
    const int kboff = (lane & 16) ? 8  : 0;     // B-frag K select

    const h16* xrow0 = X + (size_t)(m0 + mrow) * ldx;
    const h16* xrow1 = xrow0 + (size_t)16 * ldx;
    const h16* wrow[4];
#pragma unroll
    for (int g = 0; g < 4; ++g)
        wrow[g] = Wcat + (size_t)(g * AA + n0 + (lane & 15)) * K;

    v8f acc0[4] = {}, acc1[4] = {};

    // ---- part 1: x columns [0, in_dim) == weight columns [0, in_dim) ----
    for (int k0 = 0; k0 < in_dim; k0 += 32) {
        ABu a0, a1;
        a0.u[0] = *(const uint4*)(xrow0 + k0 + ksel);
        a0.u[1] = *(const uint4*)(xrow0 + k0 + 16 + ksel);
        a1.u[0] = *(const uint4*)(xrow1 + k0 + ksel);
        a1.u[1] = *(const uint4*)(xrow1 + k0 + 16 + ksel);
#pragma unroll
        for (int g = 0; g < 4; ++g) {
            ABu b;
            b.u[0] = *(const uint4*)(wrow[g] + k0 + kboff);
            b.u[1] = *(const uint4*)(wrow[g] + k0 + kboff + 16);
            acc0[g] = WMMA(a0.h, b.h, acc0[g]);
            acc1[g] = WMMA(a1.h, b.h, acc1[g]);
        }
    }
    // ---- part 2: h columns [h_read_off, +384) == weight columns [in_dim, +384) ----
    for (int k0 = 0; k0 < AA; k0 += 32) {
        ABu a0, a1;
        a0.u[0] = *(const uint4*)(xrow0 + h_read_off + k0 + ksel);
        a0.u[1] = *(const uint4*)(xrow0 + h_read_off + k0 + 16 + ksel);
        a1.u[0] = *(const uint4*)(xrow1 + h_read_off + k0 + ksel);
        a1.u[1] = *(const uint4*)(xrow1 + h_read_off + k0 + 16 + ksel);
#pragma unroll
        for (int g = 0; g < 4; ++g) {
            ABu b;
            b.u[0] = *(const uint4*)(wrow[g] + in_dim + k0 + kboff);
            b.u[1] = *(const uint4*)(wrow[g] + in_dim + k0 + kboff + 16);
            acc0[g] = WMMA(a0.h, b.h, acc0[g]);
            acc1[g] = WMMA(a1.h, b.h, acc1[g]);
        }
    }

    // ---- in-register LSTM pointwise ----
    const int rbase = (lane & 16) ? 8 : 0;
    const int ng = n0 + (lane & 15);
    const float bi = bsum[ng];
    const float bf = bsum[AA + ng];
    const float bg = bsum[2 * AA + ng];
    const float bo = bsum[3 * AA + ng];
#pragma unroll
    for (int ms = 0; ms < 2; ++ms) {
        const v8f* ai = ms ? acc1 : acc0;
#pragma unroll
        for (int r = 0; r < 8; ++r) {
            int row = m0 + ms * 16 + rbase + r;
            size_t ci = (size_t)row * AA + ng;
            float c2 = sigf(ai[1][r] + bf) * cst[ci] +
                       sigf(ai[0][r] + bi) * tanhfast(ai[2][r] + bg);
            float h2 = sigf(ai[3][r] + bo) * tanhfast(c2);
            cst[ci] = c2;
            h16 hh = (h16)h2;
            X[(size_t)row * ldx + h_write_off + ng] = hh;   // recurrent slot
            Xnext[(size_t)row * ld_next + ng] = hh;         // next consumer
        }
    }
}

// ---------------------------------------------------------------------------
// pred = sigmoid(h2 @ Wl^T + bl). Wave owns 32x16 output (2 accs, B-frag
// reused). Writes fp32 to out[:, t, :] and f16 to X0's "prev" slice.
// Grid: 32 * 24 = 768 waves = 192 blocks.
// ---------------------------------------------------------------------------
__global__ __launch_bounds__(128)
void linear_pred(const h16* __restrict__ H, const h16* __restrict__ Wl,
                 const float* __restrict__ bl,
                 float* __restrict__ out, h16* __restrict__ X0, int t) {
    const int lane = threadIdx.x & 31;
    const int wid  = blockIdx.x * 4 + (threadIdx.x >> 5);
    const int m0 = (wid / 24) * 32;
    const int n0 = (wid % 24) * 16;

    const int mrow  = lane & 15;
    const int ksel  = (lane & 16) ? 8  : 0;
    const int kboff = (lane & 16) ? 8  : 0;

    const h16* hrow0 = H  + (size_t)(m0 + mrow) * AA;
    const h16* hrow1 = hrow0 + (size_t)16 * AA;
    const h16* wrow  = Wl + (size_t)(n0 + (lane & 15)) * AA;

    v8f acc0 = {}, acc1 = {};
    for (int k0 = 0; k0 < AA; k0 += 32) {
        ABu a0, a1, b;
        a0.u[0] = *(const uint4*)(hrow0 + k0 + ksel);
        a0.u[1] = *(const uint4*)(hrow0 + k0 + 16 + ksel);
        a1.u[0] = *(const uint4*)(hrow1 + k0 + ksel);
        a1.u[1] = *(const uint4*)(hrow1 + k0 + 16 + ksel);
        b.u[0]  = *(const uint4*)(wrow + k0 + kboff);
        b.u[1]  = *(const uint4*)(wrow + k0 + kboff + 16);
        acc0 = WMMA(a0.h, b.h, acc0);
        acc1 = WMMA(a1.h, b.h, acc1);
    }

    const int rbase = (lane & 16) ? 8 : 0;
    const int n = n0 + (lane & 15);
    const float bias = bl[n];
#pragma unroll
    for (int ms = 0; ms < 2; ++ms) {
        const v8f& a = ms ? acc1 : acc0;
#pragma unroll
        for (int r = 0; r < 8; ++r) {
            int m = m0 + ms * 16 + rbase + r;
            float v = sigf(a[r] + bias);
            out[((size_t)m * LL + t) * AA + n] = v;
            X0[(size_t)m * 1280 + n] = (h16)v;   // next step's layer-0 input
        }
    }
}

// ---------------------------------------------------------------------------
extern "C" void kernel_launch(void* const* d_in, const int* in_sizes, int n_in,
                              void* d_out, int out_size, void* d_ws, size_t ws_size,
                              hipStream_t stream) {
    const float* tags = (const float*)d_in[2];
    const float* Wih[3] = {(const float*)d_in[4], (const float*)d_in[8],  (const float*)d_in[12]};
    const float* Whh[3] = {(const float*)d_in[5], (const float*)d_in[9],  (const float*)d_in[13]};
    const float* bih[3] = {(const float*)d_in[6], (const float*)d_in[10], (const float*)d_in[14]};
    const float* bhh[3] = {(const float*)d_in[7], (const float*)d_in[11], (const float*)d_in[15]};
    const float* Wl = (const float*)d_in[16];
    const float* bl = (const float*)d_in[17];
    float* out = (float*)d_out;

    // ---- carve workspace (256B aligned slices) ----
    char* p = (char*)d_ws;
    auto carve = [&p](size_t bytes) -> void* {
        void* r = (void*)p;
        p += (bytes + 255) & ~(size_t)255;
        return r;
    };
    h16*   Wcat0 = (h16*)  carve((size_t)4 * AA * (AA + TT + AA) * sizeof(h16)); // 1536x896
    h16*   Wcat1 = (h16*)  carve((size_t)4 * AA * (2 * AA) * sizeof(h16));       // 1536x768
    h16*   Wcat2 = (h16*)  carve((size_t)4 * AA * (2 * AA) * sizeof(h16));
    h16*   Wl16  = (h16*)  carve((size_t)AA * AA * sizeof(h16));
    float* bsum0 = (float*)carve((size_t)4 * AA * sizeof(float));
    float* bsum1 = (float*)carve((size_t)4 * AA * sizeof(float));
    float* bsum2 = (float*)carve((size_t)4 * AA * sizeof(float));
    h16*   X0    = (h16*)  carve((size_t)BB * 1280 * sizeof(h16));
    h16*   X1    = (h16*)  carve((size_t)BB * 1152 * sizeof(h16));
    h16*   X2    = (h16*)  carve((size_t)BB * 1152 * sizeof(h16));
    h16*   Hlin  = (h16*)  carve((size_t)BB * AA * sizeof(h16));
    float* cst   = (float*)carve((size_t)3 * BB * AA * sizeof(float));

    // ---- per-call weight conversion + state init ----
    pack_weights<<<1024, 256, 0, stream>>>(Wih[0], Whh[0], bih[0], bhh[0], Wcat0, bsum0, AA + TT);
    pack_weights<<<1024, 256, 0, stream>>>(Wih[1], Whh[1], bih[1], bhh[1], Wcat1, bsum1, AA);
    pack_weights<<<1024, 256, 0, stream>>>(Wih[2], Whh[2], bih[2], bhh[2], Wcat2, bsum2, AA);
    convert_f16<<<288, 256, 0, stream>>>(Wl, Wl16, AA * AA);
    init_state<<<1024, 256, 0, stream>>>(tags, X0, X1, X2, Hlin, cst, out);

    // ---- 49 recurrent steps ----
    for (int s = 0; s < LL - 1; ++s) {
        const int rd = s & 1, wr = (s + 1) & 1;
        lstm_layer<<<192, 128, 0, stream>>>(X0, 1280, AA + TT,
                                            (AA + TT) + rd * AA, (AA + TT) + wr * AA,
                                            Wcat0, bsum0, cst, X1, 1152);
        lstm_layer<<<192, 128, 0, stream>>>(X1, 1152, AA,
                                            AA + rd * AA, AA + wr * AA,
                                            Wcat1, bsum1, cst + (size_t)BB * AA, X2, 1152);
        lstm_layer<<<192, 128, 0, stream>>>(X2, 1152, AA,
                                            AA + rd * AA, AA + wr * AA,
                                            Wcat2, bsum2, cst + (size_t)2 * BB * AA, Hlin, AA);
        linear_pred<<<192, 128, 0, stream>>>(Hlin, Wl16, bl, out, X0, s + 1);
    }
}
